// ArBlockLSTM_10445360464476
// MI455X (gfx1250) — matrix operand order repaired and checked by reading, not compiled, not run on hardware
//
#include <hip/hip_runtime.h>
#include <hip/hip_bf16.h>
#include <stdint.h>

// ---------------------------------------------------------------------------
// AR-block LSTM for MI455X (gfx1250, wave32, WMMA).
//
// Per step: z = h @ Uh (8.6 GFLOP, bf16 WMMA, L2-resident -> compute bound)
// + x @ Wx (tiny, fp32 VALU) + b; fused gate nonlinearities; c/h update.
// 151 sequential steps, one launch per step, ping-pong bf16 h buffers.
//
// v2: register-blocked waves — each wave owns a 2x2 grid of 16x16 f32
// accumulators (32 batch x 32 units for its gate), so each K-chunk issues
// 8 global_load_b128 for 4 v_wmma_f32_16x16x32_bf16 (2 loads/WMMA instead
// of 4): A fragments reused across unit sub-tiles, B fragments reused
// across batch sub-tiles.
// ---------------------------------------------------------------------------

#define UNITS     1024
#define GATES4    (4 * UNITS)
#define FEAT      8
#define BATCH     1024
#define WARMUP_T  128
#define NBLOCKS   24
#define OUT_STEPS 192

#define BM 32   // batch tile per workgroup
#define UN 32   // unit tile per workgroup

typedef __bf16 bf16_t;
typedef __attribute__((ext_vector_type(16))) __bf16 v16bf;
typedef __attribute__((ext_vector_type(8)))  __bf16 v8bf;
typedef __attribute__((ext_vector_type(8)))  float  v8f;

__device__ __forceinline__ float fast_sigmoid(float x) {
    return 1.0f / (1.0f + __expf(-x));
}
__device__ __forceinline__ float fast_tanh(float x) {
    return 2.0f / (1.0f + __expf(-2.0f * x)) - 1.0f;
}

// Build a 16x32 bf16 A-fragment for rows [rbase..rbase+15] at K-chunk k0.
// A layout (contiguous K order, matching B): lane holds row M=lane%16; lanes 0-15 -> K 0..15,
// lanes 16-31 -> K 16..31; VGPR0-3 = first 8 K, VGPR4-7 = next 8 K.
__device__ __forceinline__ v16bf load_a_frag(const bf16_t* __restrict__ row_ptr,
                                             int k0) {
    v8bf alo = *(const v8bf*)(row_ptr + k0);
    v8bf ahi = *(const v8bf*)(row_ptr + k0 + 8);
    v16bf a;
#pragma unroll
    for (int i = 0; i < 8; ++i) { a[i] = alo[i]; a[8 + i] = ahi[i]; }
    return a;
}

// ---------------------------------------------------------------------------
// One-time: Uh [K=1024][C=4096] fp32 -> UhT [C=4096][K=1024] bf16 (LDS-tiled
// transpose so both global accesses are coalesced).
// grid (4096/32, 1024/32), block 256
// ---------------------------------------------------------------------------
__global__ __launch_bounds__(256) void prep_uhT(const float* __restrict__ Uh,
                                                bf16_t* __restrict__ UhT) {
    __shared__ float tile[32][33];
    const int cBase = blockIdx.x * 32;
    const int kBase = blockIdx.y * 32;
    const int tx = threadIdx.x & 31;
    const int ty = threadIdx.x >> 5;   // 0..7
#pragma unroll
    for (int i = 0; i < 4; ++i)
        tile[ty + i * 8][tx] =
            Uh[(size_t)(kBase + ty + i * 8) * GATES4 + cBase + tx];
    __syncthreads();
#pragma unroll
    for (int i = 0; i < 4; ++i)
        UhT[(size_t)(cBase + ty + i * 8) * UNITS + kBase + tx] =
            (bf16_t)tile[tx][ty + i * 8];
}

__global__ void zero_ws(uint32_t* __restrict__ p, size_t n_dwords) {
    size_t i = blockIdx.x * (size_t)blockDim.x + threadIdx.x;
    const size_t stride = gridDim.x * (size_t)blockDim.x;
    for (; i < n_dwords; i += stride) p[i] = 0u;
}

// ---------------------------------------------------------------------------
// One LSTM cell step. Workgroup = 128 threads = 4 waves; wave w computes the
// 32x32 z-tile of gate w (columns w*U + u0 .. +31) as a 2x2 block of 16x16
// WMMA accumulators over K=1024. Gates meet in LDS; threads then update
// 8 (b,u) cells each: c' = sig(f)*c + sig(i)*tanh(g), h' = sig(o)*tanh(c').
// grid (B/32, U/32) = (32, 32)
// ---------------------------------------------------------------------------
__global__ __launch_bounds__(128) void lstm_step(
    const bf16_t* __restrict__ hin,   // [B][U] bf16 (previous h)
    bf16_t*       __restrict__ hout,  // [B][U] bf16 (next h)
    float*        __restrict__ hf32,  // [B][U] fp32 (for Wd projection)
    float*        __restrict__ cbuf,  // [B][U] fp32 cell state (in place)
    const bf16_t* __restrict__ UhT,   // [4U][U] bf16, K contiguous per column
    const float*  __restrict__ x,     // [B][x_stride], first FEAT cols used
    const int                  x_stride,
    const float*  __restrict__ Wx,    // [FEAT][4U] fp32
    const float*  __restrict__ bias)  // [4U] fp32
{
    __shared__ float zs[4][BM][UN];   // 16 KB

    const int bm   = blockIdx.x * BM;   // batch tile base
    const int u0   = blockIdx.y * UN;   // unit tile base
    const int wave = threadIdx.x >> 5;  // == gate index (i,f,g,o)
    const int lane = threadIdx.x & 31;
    const int half = lane >> 4;         // 0: lanes 0-15, 1: lanes 16-31
    const int l16  = lane & 15;

    // A fragment base pointers for the two 16-row batch sub-tiles.
    const bf16_t* arow0 = hin + (size_t)(bm + l16)      * UNITS + half * 16;
    const bf16_t* arow1 = hin + (size_t)(bm + 16 + l16) * UNITS + half * 16;
    // B fragments (16-bit B 32x16): lane holds col N=lane%16; lanes 0-15 ->
    // K 0..15, lanes 16-31 -> K 16..31 — 32 contiguous bytes in UhT.
    const bf16_t* bcol0 =
        UhT + (size_t)(wave * UNITS + u0 + l16)      * UNITS + half * 16;
    const bf16_t* bcol1 =
        UhT + (size_t)(wave * UNITS + u0 + 16 + l16) * UNITS + half * 16;

    v8f acc00 = {}, acc01 = {}, acc10 = {}, acc11 = {};
#pragma unroll 2
    for (int k0 = 0; k0 < UNITS; k0 += 32) {
        v16bf a0 = load_a_frag(arow0, k0);
        v16bf a1 = load_a_frag(arow1, k0);
        v16bf b0 = *(const v16bf*)(bcol0 + k0);
        v16bf b1 = *(const v16bf*)(bcol1 + k0);
        acc00 = __builtin_amdgcn_wmma_f32_16x16x32_bf16(
            false, a0, false, b0, (short)0, acc00, false, false);
        acc01 = __builtin_amdgcn_wmma_f32_16x16x32_bf16(
            false, a0, false, b1, (short)0, acc01, false, false);
        acc10 = __builtin_amdgcn_wmma_f32_16x16x32_bf16(
            false, a1, false, b0, (short)0, acc10, false, false);
        acc11 = __builtin_amdgcn_wmma_f32_16x16x32_bf16(
            false, a1, false, b1, (short)0, acc11, false, false);
    }

    // C/D layout: lane holds col N=lane%16, rows M = vgpr + 8*half.
#pragma unroll
    for (int v = 0; v < 8; ++v) {
        const int m = v + half * 8;
        zs[wave][m]     [l16]      = acc00[v];
        zs[wave][m]     [l16 + 16] = acc01[v];
        zs[wave][m + 16][l16]      = acc10[v];
        zs[wave][m + 16][l16 + 16] = acc11[v];
    }
    __syncthreads();

    // Combine gates: 1024 cells / 128 threads = 8 each.
#pragma unroll
    for (int e = threadIdx.x; e < BM * UN; e += 128) {
        const int m  = e >> 5;
        const int n  = e & 31;
        const int un = u0 + n;

        float z0 = zs[0][m][n] + bias[un];
        float z1 = zs[1][m][n] + bias[UNITS + un];
        float z2 = zs[2][m][n] + bias[2 * UNITS + un];
        float z3 = zs[3][m][n] + bias[3 * UNITS + un];

        // x @ Wx contribution (K=8, fp32 VALU; Wx is L2-resident)
        const float* xr = x + (size_t)(bm + m) * x_stride;
#pragma unroll
        for (int k = 0; k < FEAT; ++k) {
            const float  xv  = xr[k];
            const float* wxr = Wx + (size_t)k * GATES4 + un;
            z0 = fmaf(xv, wxr[0],         z0);
            z1 = fmaf(xv, wxr[UNITS],     z1);
            z2 = fmaf(xv, wxr[2 * UNITS], z2);
            z3 = fmaf(xv, wxr[3 * UNITS], z3);
        }

        const float ig = fast_sigmoid(z0);
        const float fg = fast_sigmoid(z1);
        const float gg = fast_tanh(z2);
        const float og = fast_sigmoid(z3);

        const size_t idx = (size_t)(bm + m) * UNITS + un;
        const float c2 = fg * cbuf[idx] + ig * gg;
        const float h2 = og * fast_tanh(c2);
        cbuf[idx] = c2;
        hf32[idx] = h2;
        hout[idx] = (bf16_t)h2;
    }
}

// ---------------------------------------------------------------------------
// Projection: pred[b,j] = bd[j] + h[b,:] @ Wd[:,j]; also scatter this 8-wide
// block into the output. 8192 threads, K=1024 each (16 MFLOP — negligible).
// ---------------------------------------------------------------------------
__global__ __launch_bounds__(256) void proj_block(
    const float* __restrict__ h,      // [B][U] fp32
    const float* __restrict__ Wd,     // [U][FEAT]
    const float* __restrict__ bd,     // [FEAT]
    float*       __restrict__ out_blk,// d_out + block*FEAT (stride OUT_STEPS)
    float*       __restrict__ pred)   // [B][FEAT] feedback buffer
{
    const int t = blockIdx.x * blockDim.x + threadIdx.x;
    const int b = t >> 3;
    const int j = t & 7;
    const float* hr = h + (size_t)b * UNITS;
    float s = bd[j];
#pragma unroll 8
    for (int k = 0; k < UNITS; ++k)
        s = fmaf(hr[k], Wd[(size_t)k * FEAT + j], s);
    pred[(size_t)b * FEAT + j] = s;
    out_blk[(size_t)b * OUT_STEPS + j] = s;
}

// ---------------------------------------------------------------------------
// Host driver: prep + zero, 128 warm-up steps, 1 proj, 23 x (step + proj).
// All launches on `stream`; no sync, no alloc — graph-capture safe.
// ---------------------------------------------------------------------------
extern "C" void kernel_launch(void* const* d_in, const int* in_sizes, int n_in,
                              void* d_out, int out_size, void* d_ws, size_t ws_size,
                              hipStream_t stream) {
    const float* inputs = (const float*)d_in[0]; // [B][T][FEAT]
    const float* Wx     = (const float*)d_in[1]; // [FEAT][4U]
    const float* Uh     = (const float*)d_in[2]; // [U][4U]
    const float* bias   = (const float*)d_in[3]; // [4U]
    const float* Wd     = (const float*)d_in[4]; // [U][FEAT]
    const float* bd     = (const float*)d_in[5]; // [FEAT]
    float*       out    = (float*)d_out;         // [B][OUT_STEPS]

    char* ws = (char*)d_ws;
    const size_t MB = 1u << 20;
    bf16_t* UhT   = (bf16_t*)(ws);             //  8 MB: [4096][1024] bf16
    bf16_t* hb0   = (bf16_t*)(ws +  8 * MB);   //  2 MB: h ping
    bf16_t* hb1   = (bf16_t*)(ws + 10 * MB);   //  2 MB: h pong
    float*  hf32  = (float*)(ws + 12 * MB);    //  4 MB: h fp32
    float*  cbuf  = (float*)(ws + 16 * MB);    //  4 MB: cell state
    float*  pred  = (float*)(ws + 20 * MB);    // 32 KB: feedback block
    bf16_t* hb[2] = { hb0, hb1 };

    // Weight prep + zero h/c (hb0, hb1, hf32, cbuf = 12 MB contiguous).
    prep_uhT<<<dim3(GATES4 / 32, UNITS / 32), 256, 0, stream>>>(Uh, UhT);
    zero_ws<<<512, 256, 0, stream>>>((uint32_t*)(ws + 8 * MB), (12 * MB) / 4);

    const dim3 sgrid(BATCH / BM, UNITS / UN);   // 32 x 32 workgroups
    int par = 0;

    // Warm-up: x_t = inputs[:, t, :], batch row stride = T*FEAT.
    for (int t = 0; t < WARMUP_T; ++t) {
        lstm_step<<<sgrid, 128, 0, stream>>>(
            hb[par], hb[par ^ 1], hf32, cbuf, UhT,
            inputs + (size_t)t * FEAT, WARMUP_T * FEAT, Wx, bias);
        par ^= 1;
    }
    proj_block<<<(BATCH * FEAT) / 256, 256, 0, stream>>>(hf32, Wd, bd, out, pred);

    // Autoregressive blocks: feed prediction back as x (stride FEAT).
    for (int n = 1; n < NBLOCKS; ++n) {
        lstm_step<<<sgrid, 128, 0, stream>>>(
            hb[par], hb[par ^ 1], hf32, cbuf, UhT, pred, FEAT, Wx, bias);
        par ^= 1;
        proj_block<<<(BATCH * FEAT) / 256, 256, 0, stream>>>(
            hf32, Wd, bd, out + n * FEAT, pred);
    }
}
